// RNNTransducer_46600395161833
// MI455X (gfx1250) — compile-verified, matrix operand order checked by reading, not run on hardware
//
#include <hip/hip_runtime.h>
#include <hip/hip_bf16.h>

typedef __bf16 bf16_t;
typedef __attribute__((ext_vector_type(16))) __bf16 v16bf;
typedef __attribute__((ext_vector_type(8)))  float  v8f;
typedef __attribute__((ext_vector_type(4)))  float  f4_t;
typedef int i4v __attribute__((__vector_size__(4 * sizeof(int))));

#define B_   4
#define T_   256
#define U_   64
#define NC_  1024
#define IN_  80
#define EH_  320
#define DH_  512
#define OUT_ 512

// ---- CDNA5 async global->LDS copy (ASYNCcnt path), with safe fallback -------
#if defined(__HIP_DEVICE_COMPILE__) && \
    __has_builtin(__builtin_amdgcn_global_load_async_to_lds_b128)
#define HAVE_ASYNC 1
#else
#define HAVE_ASYNC 0
#endif

__device__ __forceinline__ void cp_async_b128(void* lds, const void* g) {
#if HAVE_ASYNC
  __builtin_amdgcn_global_load_async_to_lds_b128(
      (__attribute__((address_space(1))) i4v*)g,
      (__attribute__((address_space(3))) i4v*)lds, 0, 0);
#else
  *(uint4*)lds = *(const uint4*)g;
#endif
}

template <int N>
__device__ __forceinline__ void cp_wait() {
#if HAVE_ASYNC
#if __has_builtin(__builtin_amdgcn_s_wait_asynccnt)
  __builtin_amdgcn_s_wait_asynccnt((unsigned short)N);
#else
  if (N == 0) asm volatile("s_wait_asynccnt 0x0" ::: "memory");
  else        asm volatile("s_wait_asynccnt 0x1" ::: "memory");
#endif
#endif
}

__device__ __forceinline__ float sigmoidf_(float x) {
  return 1.0f / (1.0f + __expf(-x));
}

// ---------------- f32 (rows x K) -> bf16 (rows x Kpad), zero-padded ----------
__global__ void k_cvt_bf16(const float* __restrict__ src, bf16_t* __restrict__ dst,
                           int rows, int K, int Kpad) {
  int total = rows * Kpad;
  for (int i = blockIdx.x * blockDim.x + threadIdx.x; i < total;
       i += gridDim.x * blockDim.x) {
    int r = i / Kpad, k = i - r * Kpad;
    float v = (k < K) ? src[(size_t)r * K + k] : 0.0f;
    dst[i] = (bf16_t)v;
  }
}

// ---------------- embedding gather -> bf16 ----------------------------------
__global__ void k_embed(const float* __restrict__ emb, const int* __restrict__ tgt,
                        bf16_t* __restrict__ ybf) {
  int total = B_ * U_ * DH_;
  for (int i = blockIdx.x * blockDim.x + threadIdx.x; i < total;
       i += gridDim.x * blockDim.x) {
    int r = i / DH_, k = i - r * DH_;
    ybf[i] = (bf16_t)emb[(size_t)tgt[r] * DH_ + k];
  }
}

// ---------------- WMMA GEMM: C[M,N] = A[M,K] @ W[N,K]^T + bias0 + bias1 ------
// Block = 128 threads (4 waves); block tile 16(M) x 128(N); wave: two 16x16
// tiles sharing one A fragment. A staged in 64-wide K chunks, double-buffered
// in LDS via async global->LDS copies (stage c+1 while computing c).
// M % 16 == 0, N % 128 == 0, K % 32 == 0.
__global__ void k_gemm_wmma(const bf16_t* __restrict__ A, int lda,
                            const bf16_t* __restrict__ W, int ldw,
                            float* __restrict__ C, int ldc,
                            const float* __restrict__ bias0,
                            const float* __restrict__ bias1,
                            int K) {
  __shared__ bf16_t As[2][16][64];
  const int tid  = threadIdx.x;
  const int lane = tid & 31;
  const int wave = tid >> 5;                 // 0..3
  const int m0  = blockIdx.x * 16;
  const int n0a = blockIdx.y * 128 + wave * 16;
  const int n0b = n0a + 64;

  v8f acc0, acc1;
  {
    int na = n0a + (lane & 15), nb = n0b + (lane & 15);
    float bva = 0.0f, bvb = 0.0f;
    if (bias0) { bva += bias0[na]; bvb += bias0[nb]; }
    if (bias1) { bva += bias1[na]; bvb += bias1[nb]; }
#pragma unroll
    for (int v = 0; v < 8; ++v) { acc0[v] = bva; acc1[v] = bvb; }
  }

  const int srow = tid >> 3;                 // staging: 0..15
  const int scol = (tid & 7) * 8;            // staging: 0,8,..,56 (16B each)
  const int ar = lane & 15;                  // fragment row (M for A, N for B)
  const int ac = (lane >> 4) * 8;            // documented K sub-chunk pairing

  const int nchunks = (K + 63) / 64;
  {
    int klen0 = K < 64 ? K : 64;
    if (scol < klen0)
      cp_async_b128(&As[0][srow][scol], &A[(size_t)(m0 + srow) * lda + scol]);
  }

  for (int c = 0; c < nchunks; ++c) {
    const int k0 = c * 64;
    int klen = K - k0; if (klen > 64) klen = 64;
    if (c + 1 < nchunks) {
      int kn = K - (k0 + 64); if (kn > 64) kn = 64;
      if (scol < kn)
        cp_async_b128(&As[(c + 1) & 1][srow][scol],
                      &A[(size_t)(m0 + srow) * lda + k0 + 64 + scol]);
      cp_wait<1>();                          // chunk c landed; c+1 in flight
    } else {
      cp_wait<0>();
    }
    __syncthreads();
    const bf16_t (*as)[64] = As[c & 1];
    for (int kk = 0; kk < klen; kk += 32) {
      v16bf a, b0, b1;
      *(uint4*)&a       = *(const uint4*)&as[ar][kk + ac];
      *((uint4*)&a + 1) = *(const uint4*)&as[ar][kk + ac + 16];
      const bf16_t* wp0 = &W[(size_t)(n0a + ar) * ldw + k0 + kk + ac];
      const bf16_t* wp1 = &W[(size_t)(n0b + ar) * ldw + k0 + kk + ac];
      *(uint4*)&b0       = *(const uint4*)wp0;
      *((uint4*)&b0 + 1) = *(const uint4*)(wp0 + 16);
      *(uint4*)&b1       = *(const uint4*)wp1;
      *((uint4*)&b1 + 1) = *(const uint4*)(wp1 + 16);
      if (kk + 32 >= klen && k0 + 64 < K) __builtin_prefetch(wp1 + 32, 0, 1);
      acc0 = __builtin_amdgcn_wmma_f32_16x16x32_bf16(false, a, false, b0,
                                                     (short)0, acc0, false, false);
      acc1 = __builtin_amdgcn_wmma_f32_16x16x32_bf16(false, a, false, b1,
                                                     (short)0, acc1, false, false);
    }
    __syncthreads();
  }

  const int mrow = (lane >> 4) * 8;          // C layout: M = v + 8*(lane>=16)
  const int col  = lane & 15;
#pragma unroll
  for (int v = 0; v < 8; ++v) {
    C[(size_t)(m0 + mrow + v) * ldc + n0a + col] = acc0[v];
    C[(size_t)(m0 + mrow + v) * ldc + n0b + col] = acc1[v];
  }
}

// ---------------- persistent LSTM scan (one block per direction) -------------
template <int H>
__global__ void k_lstm_scan(const float* __restrict__ xg0, const bf16_t* __restrict__ whh0,
                            int off0, int rev0,
                            const float* __restrict__ xg1, const bf16_t* __restrict__ whh1,
                            int off1, int rev1,
                            float* __restrict__ out, int ostride, int seqlen) {
  constexpr int FH = 4 * H;
  __shared__ bf16_t hbf[16][H];      // padded-M hidden state (rows 4..15 zero)
  __shared__ float  gmat[4][FH];     // gate pre-activations
  __shared__ float  cst[4][H];       // cell state

  const float*  xg  = (blockIdx.x == 0) ? xg0  : xg1;
  const bf16_t* whh = (blockIdx.x == 0) ? whh0 : whh1;
  const int     off = (blockIdx.x == 0) ? off0 : off1;
  const int     rev = (blockIdx.x == 0) ? rev0 : rev1;

  const int tid  = threadIdx.x;
  const int lane = tid & 31;
  const int wave = tid >> 5;         // 0..7

  for (int i = tid; i < 16 * H; i += 256) ((bf16_t*)hbf)[i] = (bf16_t)0.0f;
  for (int i = tid; i < 4 * H; i += 256) ((float*)cst)[i] = 0.0f;
  __syncthreads();

  const int ar = lane & 15;
  const int ac = (lane >> 4) * 8;
  const int col  = lane & 15;
  const int mrow = (lane >> 4) * 8;

  for (int t = 0; t < seqlen; ++t) {
    const int tt = rev ? (seqlen - 1 - t) : t;
    for (int nt = wave; nt < FH / 16; nt += 8) {
      const int n0 = nt * 16;
      v8f acc;
#pragma unroll
      for (int v = 0; v < 8; ++v) {
        int m = mrow + v;
        acc[v] = (m < 4) ? xg[((size_t)m * seqlen + tt) * FH + n0 + col] : 0.0f;
      }
      for (int k0 = 0; k0 < H; k0 += 32) {
        v16bf a, b;
        *(uint4*)&a       = *(const uint4*)&hbf[ar][k0 + ac];
        *((uint4*)&a + 1) = *(const uint4*)&hbf[ar][k0 + ac + 16];
        const bf16_t* wp = &whh[(size_t)(n0 + ar) * H + k0 + ac];
        *(uint4*)&b       = *(const uint4*)wp;
        *((uint4*)&b + 1) = *(const uint4*)(wp + 16);
        acc = __builtin_amdgcn_wmma_f32_16x16x32_bf16(false, a, false, b,
                                                      (short)0, acc, false, false);
      }
#pragma unroll
      for (int v = 0; v < 8; ++v) {
        int m = mrow + v;
        if (m < 4) gmat[m][n0 + col] = acc[v];
      }
    }
    __syncthreads();
    for (int i = tid; i < 4 * H; i += 256) {
      int bb = i / H, h = i - bb * H;
      float ig = sigmoidf_(gmat[bb][h]);
      float fg = sigmoidf_(gmat[bb][H + h]);
      float gg = tanhf(gmat[bb][2 * H + h]);
      float og = sigmoidf_(gmat[bb][3 * H + h]);
      float c  = fg * cst[bb][h] + ig * gg;
      cst[bb][h] = c;
      float hn = og * tanhf(c);
      hbf[bb][h] = (bf16_t)hn;
      out[((size_t)bb * seqlen + tt) * ostride + off + h] = hn;
    }
    __syncthreads();
  }
}

// -------- joint: out[b,t,u,:] = ej[b,t,:] + dj[b,u,:]  (256 MB NT stream) ----
// grid = (U*NC/4/256, B*T); all index math is shifts (powers of two).
__global__ void k_joint(const f4_t* __restrict__ ej, const f4_t* __restrict__ dj,
                        f4_t* __restrict__ out) {
  const int NC4 = NC_ / 4;                               // 256
  int i2 = blockIdx.x * blockDim.x + threadIdx.x;        // 0 .. U*NC4-1
  int bt = blockIdx.y;                                   // b*T + t
  int u  = i2 >> 8;                                      // / NC4
  int n4 = i2 & (NC4 - 1);
  int b  = bt >> 8;                                      // / T_
  f4_t e = ej[(size_t)bt * NC4 + n4];
  f4_t d = dj[((size_t)b * U_ + u) * NC4 + n4];
  __builtin_nontemporal_store(e + d, &out[((size_t)bt * U_ + u) * NC4 + n4]);
}

extern "C" void kernel_launch(void* const* d_in, const int* in_sizes, int n_in,
                              void* d_out, int out_size, void* d_ws, size_t ws_size,
                              hipStream_t stream) {
  (void)in_sizes; (void)n_in; (void)out_size; (void)ws_size;
  const float* inputs  = (const float*)d_in[0];
  const int*   targets = (const int*)d_in[1];
  const float *e_wih[4][2], *e_whh[4][2], *e_bih[4][2], *e_bhh[4][2];
  int idx = 4;
  for (int l = 0; l < 4; ++l)
    for (int d = 0; d < 2; ++d) {
      e_wih[l][d] = (const float*)d_in[idx++];
      e_whh[l][d] = (const float*)d_in[idx++];
      e_bih[l][d] = (const float*)d_in[idx++];
      e_bhh[l][d] = (const float*)d_in[idx++];
    }
  const float* e_pw  = (const float*)d_in[36];
  const float* e_pb  = (const float*)d_in[37];
  const float* emb   = (const float*)d_in[38];
  const float* d_wih = (const float*)d_in[39];
  const float* d_whh = (const float*)d_in[40];
  const float* d_bih = (const float*)d_in[41];
  const float* d_bhh = (const float*)d_in[42];
  const float* d_pw  = (const float*)d_in[43];
  const float* d_pb  = (const float*)d_in[44];
  const float* fc_w  = (const float*)d_in[45];

  char* ws = (char*)d_ws;
  size_t off = 0;
  auto alloc = [&](size_t bytes) -> void* {
    void* p = ws + off;
    off = (off + bytes + 255) & ~(size_t)255;
    return p;
  };
  const int MT = B_ * T_;   // 1024
  const int MU = B_ * U_;   // 256

  bf16_t* Xbf    = (bf16_t*)alloc((size_t)MT * 640 * 2);
  bf16_t* wihbf  = (bf16_t*)alloc((size_t)1280 * 640 * 2);
  bf16_t* whhf   = (bf16_t*)alloc((size_t)1280 * 320 * 2);
  bf16_t* whhb   = (bf16_t*)alloc((size_t)1280 * 320 * 2);
  float*  xgf    = (float*)alloc((size_t)MT * 1280 * 4);
  float*  xgb    = (float*)alloc((size_t)MT * 1280 * 4);
  float*  Xn     = (float*)alloc((size_t)MT * 640 * 4);
  float*  enc    = (float*)alloc((size_t)MT * 512 * 4);
  bf16_t* encbf  = (bf16_t*)alloc((size_t)MT * 512 * 2);
  bf16_t* epwbf  = (bf16_t*)alloc((size_t)512 * 640 * 2);
  bf16_t* fcbf   = (bf16_t*)alloc((size_t)1024 * 1024 * 2);
  float*  ej     = (float*)alloc((size_t)MT * 1024 * 4);
  bf16_t* ybf    = (bf16_t*)alloc((size_t)MU * 512 * 2);
  bf16_t* dwihbf = (bf16_t*)alloc((size_t)2048 * 512 * 2);
  bf16_t* dwhhbf = (bf16_t*)alloc((size_t)2048 * 512 * 2);
  float*  dxg    = (float*)alloc((size_t)MU * 2048 * 4);
  float*  dh     = (float*)alloc((size_t)MU * 512 * 4);
  bf16_t* dhbf   = (bf16_t*)alloc((size_t)MU * 512 * 2);
  bf16_t* dpwbf  = (bf16_t*)alloc((size_t)512 * 512 * 2);
  float*  dp     = (float*)alloc((size_t)MU * 512 * 4);
  bf16_t* dpbf   = (bf16_t*)alloc((size_t)MU * 512 * 2);
  float*  dj     = (float*)alloc((size_t)MU * 1024 * 4);

  auto cvt = [&](const float* s, bf16_t* dptr, int rows, int K, int Kpad) {
    int total = rows * Kpad;
    int blocks = (total + 255) / 256;
    if (blocks > 2048) blocks = 2048;
    hipLaunchKernelGGL(k_cvt_bf16, dim3(blocks), dim3(256), 0, stream,
                       s, dptr, rows, K, Kpad);
  };
  auto gemm = [&](const bf16_t* A, int lda, const bf16_t* W, int ldw,
                  float* C, int ldc, const float* b0, const float* b1,
                  int M, int N, int K) {
    hipLaunchKernelGGL(k_gemm_wmma, dim3(M / 16, N / 128), dim3(128), 0, stream,
                       A, lda, W, ldw, C, ldc, b0, b1, K);
  };

  // ---------------- encoder ----------------
  cvt(inputs, Xbf, MT, IN_, 96);
  int Kp = 96;
  for (int l = 0; l < 4; ++l) {
    int inK = (l == 0) ? IN_ : 640;
    cvt(e_wih[l][0], wihbf, 1280, inK, Kp);
    gemm(Xbf, Kp, wihbf, Kp, xgf, 1280, e_bih[l][0], e_bhh[l][0], MT, 1280, Kp);
    cvt(e_wih[l][1], wihbf, 1280, inK, Kp);       // stream-ordered reuse
    gemm(Xbf, Kp, wihbf, Kp, xgb, 1280, e_bih[l][1], e_bhh[l][1], MT, 1280, Kp);
    cvt(e_whh[l][0], whhf, 1280, 320, 320);
    cvt(e_whh[l][1], whhb, 1280, 320, 320);
    hipLaunchKernelGGL((k_lstm_scan<320>), dim3(2), dim3(256), 0, stream,
                       xgf, whhf, 0, 0, xgb, whhb, 320, 1, Xn, 640, T_);
    cvt(Xn, Xbf, MT, 640, 640);
    Kp = 640;
  }
  cvt(e_pw, epwbf, 512, 640, 640);
  gemm(Xbf, 640, epwbf, 640, enc, 512, e_pb, nullptr, MT, 512, 640);
  cvt(enc, encbf, MT, 512, 512);
  cvt(fc_w, fcbf, 1024, 1024, 1024);
  gemm(encbf, 512, fcbf, 1024, ej, 1024, nullptr, nullptr, MT, 1024, 512);

  // ---------------- decoder ----------------
  {
    int blocks = (MU * 512 + 255) / 256;
    hipLaunchKernelGGL(k_embed, dim3(blocks), dim3(256), 0, stream,
                       emb, targets, ybf);
  }
  cvt(d_wih, dwihbf, 2048, 512, 512);
  gemm(ybf, 512, dwihbf, 512, dxg, 2048, d_bih, d_bhh, MU, 2048, 512);
  cvt(d_whh, dwhhbf, 2048, 512, 512);
  hipLaunchKernelGGL((k_lstm_scan<512>), dim3(1), dim3(256), 0, stream,
                     dxg, dwhhbf, 0, 0, dxg, dwhhbf, 0, 0, dh, 512, U_);
  cvt(dh, dhbf, MU, 512, 512);
  cvt(d_pw, dpwbf, 512, 512, 512);
  gemm(dhbf, 512, dpwbf, 512, dp, 512, d_pb, nullptr, MU, 512, 512);
  cvt(dp, dpbf, MU, 512, 512);
  gemm(dpbf, 512, fcbf + 512, 1024, dj, 1024, nullptr, nullptr, MU, 1024, 512);

  // ---------------- joint (bandwidth-bound: 256 MB NT stream) ----------------
  hipLaunchKernelGGL(k_joint, dim3((U_ * NC_ / 4) / 256, MT), dim3(256), 0, stream,
                     (const f4_t*)ej, (const f4_t*)dj, (f4_t*)d_out);
}
